// CAL_GAT_64115271794809
// MI455X (gfx1250) — compile-verified, matrix operand order checked by reading, not run on hardware
//
#include <hip/hip_runtime.h>

// ---------------- problem constants (from reference) ----------------
#define N_NODES 100000
#define N_EDGES 800000
#define E2      (N_EDGES + N_NODES)   // edges + self loops
#define F_IN    128
#define HEADS   8
#define HID     8
#define HD      64                    // HEADS*HID
#define G_GRAPHS 128
#define C_CLS   10
#define SLOPE   0.2f

typedef __attribute__((ext_vector_type(16))) __bf16          v16bf;
typedef __attribute__((ext_vector_type(16))) unsigned short  v16u;
typedef __attribute__((ext_vector_type(8)))  float           v8f;

// fp32 -> bf16 (round-to-nearest-even)
__device__ __forceinline__ unsigned short f2bf(float f) {
  unsigned u = __float_as_uint(f);
  u += 0x7FFFu + ((u >> 16) & 1u);
  return (unsigned short)(u >> 16);
}

// order-preserving float<->uint mapping for atomic segment-max
__device__ __forceinline__ unsigned fmap(float f) {
  unsigned u = __float_as_uint(f);
  return (u & 0x80000000u) ? ~u : (u | 0x80000000u);
}
__device__ __forceinline__ float funmap(unsigned u) {
  u = (u & 0x80000000u) ? (u & 0x7FFFFFFFu) : ~u;
  return __uint_as_float(u);
}

__device__ __forceinline__ float lrelu(float x) { return x > 0.f ? x : SLOPE * x; }

// K index pattern for 16-bit A/B fragments (ISA 7.12.2): VGPR v holds K pair
// {2v,2v+1} (v<4) or {16+2(v-4), ...}; upper 16 lanes add +8.
__device__ __forceinline__ int kpat(int v) { return (v < 4) ? 2 * v : 16 + 2 * (v - 4); }

// ---------------- fill kernels ----------------
__global__ void k_fill_f32(float* p, float v, long long n) {
  for (long long i = (long long)blockIdx.x * blockDim.x + threadIdx.x; i < n;
       i += (long long)gridDim.x * blockDim.x) p[i] = v;
}
__global__ void k_fill_u32(unsigned* p, unsigned v, long long n) {
  for (long long i = (long long)blockIdx.x * blockDim.x + threadIdx.x; i < n;
       i += (long long)gridDim.x * blockDim.x) p[i] = v;
}

// ---------------- bf16 WMMA GEMM:  C[M x 64] = A[M x K] @ B[K x 64] ----------------
// B (weights, <=128x64) is staged once per block into LDS as bf16; one wave owns a
// 16-row stripe and all four 16-col tiles (A fragment reused 4x).
__global__ void k_gemm_bf16_nc64(const float* __restrict__ A,
                                 const float* __restrict__ B,
                                 float* __restrict__ C,
                                 int M, int K) {
  __shared__ unsigned short swb[F_IN * HD];          // up to 128x64 bf16 = 16KB
  for (int i = threadIdx.x; i < K * HD; i += blockDim.x) swb[i] = f2bf(B[i]);
  __syncthreads();

  const int wave = threadIdx.x >> 5;
  const int lane = threadIdx.x & 31;
  const int mt   = blockIdx.x * (blockDim.x >> 5) + wave;
  if (mt >= (M >> 4)) return;               // wave-uniform; after the block sync
  const int hlf  = lane >> 4;
  const int lrow = lane & 15;
  const int row  = (mt << 4) + lrow;
  const float* Arow = A + (size_t)row * K;

  v8f acc[4] = {v8f{}, v8f{}, v8f{}, v8f{}};

  for (int k0 = 0; k0 < K; k0 += 32) {
    __builtin_prefetch(Arow + k0 + 32, 0, 1);        // global_prefetch next K chunk
    v16u au;
#pragma unroll
    for (int v = 0; v < 8; ++v) {
      int kk = k0 + hlf * 8 + kpat(v);
      au[2 * v]     = f2bf(Arow[kk]);
      au[2 * v + 1] = f2bf(Arow[kk + 1]);
    }
    v16bf af = __builtin_bit_cast(v16bf, au);
#pragma unroll
    for (int nt = 0; nt < 4; ++nt) {
      v16u bu;
#pragma unroll
      for (int v = 0; v < 8; ++v) {
        int kk = k0 + hlf * 8 + kpat(v);
        bu[2 * v]     = swb[kk * HD + nt * 16 + lrow];
        bu[2 * v + 1] = swb[(kk + 1) * HD + nt * 16 + lrow];
      }
      v16bf bm = __builtin_bit_cast(v16bf, bu);
      acc[nt] = __builtin_amdgcn_wmma_f32_16x16x32_bf16(
          false, af, false, bm, (short)0, acc[nt], false, false);
    }
  }
  // C/D layout: VGPR v -> M = v + 8*half, N = lane&15
#pragma unroll
  for (int nt = 0; nt < 4; ++nt) {
    int col = nt * 16 + lrow;
#pragma unroll
    for (int v = 0; v < 8; ++v) {
      int r = (mt << 4) + v + 8 * hlf;
      C[(size_t)r * HD + col] = acc[nt][v];
    }
  }
}

// ---------------- GAT pieces ----------------
__global__ void k_attn_dots(const float* __restrict__ H,
                            const float* __restrict__ a_s,
                            const float* __restrict__ a_d,
                            float* __restrict__ es, float* __restrict__ ed) {
  int n = blockIdx.x * blockDim.x + threadIdx.x;
  if (n >= N_NODES) return;
  const float* h = H + (size_t)n * HD;
#pragma unroll
  for (int hh = 0; hh < HEADS; ++hh) {
    float s = 0.f, d = 0.f;
#pragma unroll
    for (int k = 0; k < HID; ++k) {
      float v = h[hh * HID + k];
      s += v * a_s[hh * HID + k];
      d += v * a_d[hh * HID + k];
    }
    es[n * HEADS + hh] = s;
    ed[n * HEADS + hh] = d;
  }
}

__global__ void k_edge_max(const int* __restrict__ src, const int* __restrict__ dst,
                           const float* __restrict__ es, const float* __restrict__ ed,
                           unsigned* __restrict__ m) {
  int idx = blockIdx.x * blockDim.x + threadIdx.x;
  if (idx >= E2 * HEADS) return;
  int e = idx >> 3, hh = idx & 7;
  int s, d;
  if (e < N_EDGES) { s = src[e]; d = dst[e]; } else { s = d = e - N_EDGES; }
  float l = lrelu(es[s * HEADS + hh] + ed[d * HEADS + hh]);
  atomicMax(&m[d * HEADS + hh], fmap(l));
}

__global__ void k_edge_expsum(const int* __restrict__ src, const int* __restrict__ dst,
                              const float* __restrict__ es, const float* __restrict__ ed,
                              const unsigned* __restrict__ m, float* __restrict__ den) {
  int idx = blockIdx.x * blockDim.x + threadIdx.x;
  if (idx >= E2 * HEADS) return;
  int e = idx >> 3, hh = idx & 7;
  int s, d;
  if (e < N_EDGES) { s = src[e]; d = dst[e]; } else { s = d = e - N_EDGES; }
  float l = lrelu(es[s * HEADS + hh] + ed[d * HEADS + hh]);
  atomicAdd(&den[d * HEADS + hh], expf(l - funmap(m[d * HEADS + hh])));
}

__global__ void k_edge_agg(const int* __restrict__ src, const int* __restrict__ dst,
                           const float* __restrict__ es, const float* __restrict__ ed,
                           const unsigned* __restrict__ m, const float* __restrict__ den,
                           const float* __restrict__ H, float* __restrict__ out) {
  long long idx = (long long)blockIdx.x * blockDim.x + threadIdx.x;
  if (idx >= (long long)E2 * HD) return;
  int e = (int)(idx >> 6), j = (int)(idx & 63), hh = j >> 3;
  int s, d;
  if (e < N_EDGES) { s = src[e]; d = dst[e]; } else { s = d = e - N_EDGES; }
  float l = lrelu(es[s * HEADS + hh] + ed[d * HEADS + hh]);
  float ex = expf(l - funmap(m[d * HEADS + hh]));
  float alpha = ex / (den[d * HEADS + hh] + 1e-16f);
  atomicAdd(&out[(size_t)d * HD + j], H[(size_t)s * HD + j] * alpha);
}

// mode 0: +bias then ELU; mode 1: +bias only
__global__ void k_bias_act(float* __restrict__ H, const float* __restrict__ b, int mode) {
  long long idx = (long long)blockIdx.x * blockDim.x + threadIdx.x;
  if (idx >= (long long)N_NODES * HD) return;
  int j = (int)(idx & 63);
  float v = H[idx] + b[j];
  if (mode == 0) v = v > 0.f ? v : (expf(v) - 1.f);
  H[idx] = v;
}

// ---------------- node attention MLP (small: scalar) ----------------
__global__ void k_node_attn(const float* __restrict__ H,
                            const float* __restrict__ w1, const float* __restrict__ b1,
                            const float* __restrict__ w2, const float* __restrict__ b2,
                            float* __restrict__ ac, float* __restrict__ at) {
  __shared__ float sw1[HD * 32];
  __shared__ float sw2[32 * 2];
  for (int i = threadIdx.x; i < HD * 32; i += blockDim.x) sw1[i] = w1[i];
  for (int i = threadIdx.x; i < 32 * 2; i += blockDim.x) sw2[i] = w2[i];
  __syncthreads();
  int n = blockIdx.x * blockDim.x + threadIdx.x;
  if (n >= N_NODES) return;
  const float* h = H + (size_t)n * HD;
  float z0 = b2[0], z1 = b2[1];
  for (int k = 0; k < 32; ++k) {
    float t = b1[k];
    for (int i = 0; i < HD; ++i) t += h[i] * sw1[i * 32 + k];
    t = t > 0.f ? t : 0.f;
    z0 += t * sw2[k * 2 + 0];
    z1 += t * sw2[k * 2 + 1];
  }
  float mx = fmaxf(z0, z1);
  float e0 = expf(z0 - mx), e1 = expf(z1 - mx);
  float inv = 1.f / (e0 + e1);
  ac[n] = e0 * inv;
  at[n] = e1 * inv;
}

// ---------------- edge attention MLP via WMMA (E x 128 @ 128 x 64 -> 64 -> 2) ----
// One wave = one 16-edge tile. A-fragment rows are gathered H[src]/H[dst] rows
// (k<64 -> src half, k>=64 -> dst half) so the 409MB edge-feature matrix is never
// materialized. Layer-2 (64->2) is a per-lane partial dot + shfl_xor butterfly
// across the 16-lane half that owns each edge row.
__global__ void k_edge_attn_wmma(const float* __restrict__ H,
                                 const int* __restrict__ src, const int* __restrict__ dst,
                                 const float* __restrict__ w1, const float* __restrict__ b1,
                                 const float* __restrict__ w2, const float* __restrict__ b2,
                                 float* __restrict__ bc, float* __restrict__ bt) {
  __shared__ unsigned short sw1[2 * HD * HD];        // 128x64 bf16 = 16KB
  for (int i = threadIdx.x; i < 2 * HD * HD; i += blockDim.x) sw1[i] = f2bf(w1[i]);
  __syncthreads();

  const int wave = threadIdx.x >> 5;
  const int lane = threadIdx.x & 31;
  const int et   = blockIdx.x * (blockDim.x >> 5) + wave;
  if (et >= N_EDGES / 16) return;                    // wave-uniform; after sync
  const int hlf  = lane >> 4;
  const int lrow = lane & 15;
  const int er   = et * 16 + lrow;
  const float* hs = H + (size_t)src[er] * HD;
  const float* hd = H + (size_t)dst[er] * HD;

  v8f acc[4] = {v8f{}, v8f{}, v8f{}, v8f{}};

#pragma unroll
  for (int k0 = 0; k0 < 2 * HD; k0 += 32) {
    const float* Abase = (k0 < HD) ? (hs + k0) : (hd + (k0 - HD));
    v16u au;
#pragma unroll
    for (int v = 0; v < 8; ++v) {
      int kk = hlf * 8 + kpat(v);
      au[2 * v]     = f2bf(Abase[kk]);
      au[2 * v + 1] = f2bf(Abase[kk + 1]);
    }
    v16bf af = __builtin_bit_cast(v16bf, au);
#pragma unroll
    for (int nt = 0; nt < 4; ++nt) {
      v16u bu;
#pragma unroll
      for (int v = 0; v < 8; ++v) {
        int kk = k0 + hlf * 8 + kpat(v);
        bu[2 * v]     = sw1[kk * HD + nt * 16 + lrow];
        bu[2 * v + 1] = sw1[(kk + 1) * HD + nt * 16 + lrow];
      }
      v16bf bm = __builtin_bit_cast(v16bf, bu);
      acc[nt] = __builtin_amdgcn_wmma_f32_16x16x32_bf16(
          false, af, false, bm, (short)0, acc[nt], false, false);
    }
  }

  // bias + relu + layer-2 partials: this lane holds hidden[N = nt*16+lrow] for
  // edges M = v + 8*hlf.
  float p0[8] = {0, 0, 0, 0, 0, 0, 0, 0};
  float p1[8] = {0, 0, 0, 0, 0, 0, 0, 0};
#pragma unroll
  for (int nt = 0; nt < 4; ++nt) {
    int ncol = nt * 16 + lrow;
    float bb  = b1[ncol];
    float w20 = w2[ncol * 2 + 0];
    float w21 = w2[ncol * 2 + 1];
#pragma unroll
    for (int v = 0; v < 8; ++v) {
      float t = acc[nt][v] + bb;
      t = t > 0.f ? t : 0.f;
      p0[v] += t * w20;
      p1[v] += t * w21;
    }
  }
  // reduce over the 16 lanes of this half (xor masks < 16 stay within the half)
#pragma unroll
  for (int msk = 1; msk < 16; msk <<= 1) {
#pragma unroll
    for (int v = 0; v < 8; ++v) {
      p0[v] += __shfl_xor(p0[v], msk, 32);
      p1[v] += __shfl_xor(p1[v], msk, 32);
    }
  }
  if (lrow == 0) {
    float bz0 = b2[0], bz1 = b2[1];
#pragma unroll
    for (int v = 0; v < 8; ++v) {
      int e = et * 16 + v + 8 * hlf;
      float z0 = p0[v] + bz0, z1 = p1[v] + bz1;
      float mx = fmaxf(z0, z1);
      float e0 = expf(z0 - mx), e1 = expf(z1 - mx);
      float inv = 1.f / (e0 + e1);
      bc[e] = e0 * inv;
      bt[e] = e1 * inv;
    }
  }
}

__global__ void k_scale_rows(const float* __restrict__ H, const float* __restrict__ a,
                             float* __restrict__ out) {
  long long idx = (long long)blockIdx.x * blockDim.x + threadIdx.x;
  if (idx >= (long long)N_NODES * HD) return;
  out[idx] = H[idx] * a[idx >> 6];
}

// ---------------- GCN pieces ----------------
__global__ void k_deg(const int* __restrict__ dst, const float* __restrict__ beta,
                      float* __restrict__ deg) {
  int e = blockIdx.x * blockDim.x + threadIdx.x;
  if (e >= E2) return;
  int d; float w;
  if (e < N_EDGES) { d = dst[e]; w = beta[e]; } else { d = e - N_EDGES; w = 1.f; }
  atomicAdd(&deg[d], w);
}
__global__ void k_dis(float* __restrict__ deg) {
  int n = blockIdx.x * blockDim.x + threadIdx.x;
  if (n >= N_NODES) return;
  float d = deg[n];
  deg[n] = d > 0.f ? rsqrtf(fmaxf(d, 1e-20f)) : 0.f;
}
__global__ void k_gcn_scatter(const int* __restrict__ src, const int* __restrict__ dst,
                              const float* __restrict__ beta, const float* __restrict__ dis,
                              const float* __restrict__ P, float* __restrict__ out) {
  long long idx = (long long)blockIdx.x * blockDim.x + threadIdx.x;
  if (idx >= (long long)E2 * HD) return;
  int e = (int)(idx >> 6), j = (int)(idx & 63);
  int s, d; float w;
  if (e < N_EDGES) { s = src[e]; d = dst[e]; w = beta[e]; } else { s = d = e - N_EDGES; w = 1.f; }
  float norm = dis[s] * w * dis[d];
  atomicAdd(&out[(size_t)d * HD + j], P[(size_t)s * HD + j] * norm);
}

// ---------------- readout + heads ----------------
__global__ void k_readout(const float* __restrict__ P, const float* __restrict__ b,
                          const int* __restrict__ batch, float* __restrict__ hG) {
  long long idx = (long long)blockIdx.x * blockDim.x + threadIdx.x;
  if (idx >= (long long)N_NODES * HD) return;
  int n = (int)(idx >> 6), j = (int)(idx & 63);
  atomicAdd(&hG[(size_t)batch[n] * HD + j], P[idx] + b[j]);
}

__global__ void k_mlp_head(const float* __restrict__ hG,
                           const float* __restrict__ w1, const float* __restrict__ b1,
                           const float* __restrict__ w2, const float* __restrict__ b2,
                           float* __restrict__ z) {
  int g = blockIdx.x * blockDim.x + threadIdx.x;
  if (g >= G_GRAPHS) return;
  const float* h = hG + (size_t)g * HD;
  float t[32];
  for (int k = 0; k < 32; ++k) {
    float s = b1[k];
    for (int i = 0; i < HD; ++i) s += h[i] * w1[i * 32 + k];
    t[k] = s > 0.f ? s : 0.f;
  }
  for (int c = 0; c < C_CLS; ++c) {
    float s = b2[c];
    for (int k = 0; k < 32; ++k) s += t[k] * w2[k * C_CLS + c];
    z[g * C_CLS + c] = s;
  }
}

__global__ void k_mlp_comb(const float* __restrict__ hGc, const float* __restrict__ hGt,
                           const float* __restrict__ w1, const float* __restrict__ b1,
                           const float* __restrict__ w2, const float* __restrict__ b2,
                           float* __restrict__ z) {
  int g = blockIdx.x * blockDim.x + threadIdx.x;
  if (g >= G_GRAPHS) return;
  float t[HD];
  for (int k = 0; k < HD; ++k) {
    float s = b1[k];
    for (int i = 0; i < HD; ++i) s += hGc[(size_t)g * HD + i] * w1[i * HD + k];
    for (int i = 0; i < HD; ++i) s += hGt[(size_t)g * HD + i] * w1[(HD + i) * HD + k];
    t[k] = s > 0.f ? s : 0.f;
  }
  for (int c = 0; c < C_CLS; ++c) {
    float s = b2[c];
    for (int k = 0; k < HD; ++k) s += t[k] * w2[k * C_CLS + c];
    z[g * C_CLS + c] = s;
  }
}

// ---------------- launcher ----------------
static inline unsigned gsz(long long n, int b) { return (unsigned)((n + b - 1) / b); }

extern "C" void kernel_launch(void* const* d_in, const int* in_sizes, int n_in,
                              void* d_out, int out_size, void* d_ws, size_t ws_size,
                              hipStream_t stream) {
  (void)in_sizes; (void)n_in; (void)out_size; (void)ws_size;
  const float* x     = (const float*)d_in[0];
  const int*   ei    = (const int*)d_in[1];
  const int*   src   = ei;
  const int*   dst   = ei + N_EDGES;
  const int*   batch = (const int*)d_in[2];
  const float* W1  = (const float*)d_in[3];
  const float* as1 = (const float*)d_in[4];
  const float* ad1 = (const float*)d_in[5];
  const float* b1  = (const float*)d_in[6];
  const float* W2  = (const float*)d_in[7];
  const float* as2 = (const float*)d_in[8];
  const float* ad2 = (const float*)d_in[9];
  const float* b2  = (const float*)d_in[10];
  const float* na_w1 = (const float*)d_in[11];
  const float* na_b1 = (const float*)d_in[12];
  const float* na_w2 = (const float*)d_in[13];
  const float* na_b2 = (const float*)d_in[14];
  const float* ea_w1 = (const float*)d_in[15];
  const float* ea_b1 = (const float*)d_in[16];
  const float* ea_w2 = (const float*)d_in[17];
  const float* ea_b2 = (const float*)d_in[18];
  const float* gc_w = (const float*)d_in[19];
  const float* gc_b = (const float*)d_in[20];
  const float* gt_w = (const float*)d_in[21];
  const float* gt_b = (const float*)d_in[22];
  const float* rc_w = (const float*)d_in[23];
  const float* rc_b = (const float*)d_in[24];
  const float* rt_w = (const float*)d_in[25];
  const float* rt_b = (const float*)d_in[26];
  const float* cc_w1 = (const float*)d_in[27];
  const float* cc_b1 = (const float*)d_in[28];
  const float* cc_w2 = (const float*)d_in[29];
  const float* cc_b2 = (const float*)d_in[30];
  const float* ct_w1 = (const float*)d_in[31];
  const float* ct_b1 = (const float*)d_in[32];
  const float* ct_w2 = (const float*)d_in[33];
  const float* ct_b2 = (const float*)d_in[34];
  const float* cm_w1 = (const float*)d_in[35];
  const float* cm_b1 = (const float*)d_in[36];
  const float* cm_w2 = (const float*)d_in[37];
  const float* cm_b2 = (const float*)d_in[38];

  float* out = (float*)d_out;
  float* zc  = out;
  float* zt  = out + G_GRAPHS * C_CLS;
  float* zp  = out + 2 * G_GRAPHS * C_CLS;
  float* hGc = out + 3 * G_GRAPHS * C_CLS;
  float* hGt = hGc + G_GRAPHS * HD;

  // workspace layout (floats)
  const size_t NF = (size_t)N_NODES * HD;
  float* ws = (float*)d_ws;
  float* B0 = ws;            // Hpre (L1,L2), later Hc
  float* B1 = B0 + NF;       // H1 agg/out, later Ht
  float* B2 = B1 + NF;       // H agg/out, later Gt
  float* B3 = B2 + NF;       // GEMM scratch (P)
  float* B4 = B3 + NF;       // Gc
  float* ES = B4 + NF;                                    // N*8, later deg/dis (c)
  float* ED = ES + (size_t)N_NODES * HEADS;               // N*8, later deg/dis (t)
  unsigned* Mb = (unsigned*)(ED + (size_t)N_NODES * HEADS);
  float* DEN = (float*)(Mb + (size_t)N_NODES * HEADS);
  float* AC = DEN + (size_t)N_NODES * HEADS;
  float* AT = AC + N_NODES;
  float* BC = AT + N_NODES;
  float* BT = BC + N_EDGES;

  const int T = 256;
  const unsigned gemmGrid = gsz(N_NODES / 16, T / 32);     // 8 waves/block
  const unsigned fillGrid = 2048;
  const long long NHD  = (long long)N_NODES * HD;
  const long long E2H  = (long long)E2 * HEADS;
  const long long E2HD = (long long)E2 * HD;

  // ================= GAT layer 1 =================
  k_gemm_bf16_nc64<<<gemmGrid, T, 0, stream>>>(x, W1, B0, N_NODES, F_IN);
  k_attn_dots<<<gsz(N_NODES, T), T, 0, stream>>>(B0, as1, ad1, ES, ED);
  k_fill_u32<<<fillGrid, T, 0, stream>>>(Mb, 0u, (long long)N_NODES * HEADS);
  k_fill_f32<<<fillGrid, T, 0, stream>>>(DEN, 0.f, (long long)N_NODES * HEADS);
  k_fill_f32<<<fillGrid, T, 0, stream>>>(B1, 0.f, NHD);
  k_edge_max<<<gsz(E2H, T), T, 0, stream>>>(src, dst, ES, ED, Mb);
  k_edge_expsum<<<gsz(E2H, T), T, 0, stream>>>(src, dst, ES, ED, Mb, DEN);
  k_edge_agg<<<gsz(E2HD, T), T, 0, stream>>>(src, dst, ES, ED, Mb, DEN, B0, B1);
  k_bias_act<<<gsz(NHD, T), T, 0, stream>>>(B1, b1, 0);    // +b1, ELU -> H1

  // ================= GAT layer 2 =================
  k_gemm_bf16_nc64<<<gemmGrid, T, 0, stream>>>(B1, W2, B0, N_NODES, HD);
  k_attn_dots<<<gsz(N_NODES, T), T, 0, stream>>>(B0, as2, ad2, ES, ED);
  k_fill_u32<<<fillGrid, T, 0, stream>>>(Mb, 0u, (long long)N_NODES * HEADS);
  k_fill_f32<<<fillGrid, T, 0, stream>>>(DEN, 0.f, (long long)N_NODES * HEADS);
  k_fill_f32<<<fillGrid, T, 0, stream>>>(B2, 0.f, NHD);
  k_edge_max<<<gsz(E2H, T), T, 0, stream>>>(src, dst, ES, ED, Mb);
  k_edge_expsum<<<gsz(E2H, T), T, 0, stream>>>(src, dst, ES, ED, Mb, DEN);
  k_edge_agg<<<gsz(E2HD, T), T, 0, stream>>>(src, dst, ES, ED, Mb, DEN, B0, B2);
  k_bias_act<<<gsz(NHD, T), T, 0, stream>>>(B2, b2, 1);    // +b2 -> H

  // ================= node / edge attention =================
  k_node_attn<<<gsz(N_NODES, T), T, 0, stream>>>(B2, na_w1, na_b1, na_w2, na_b2, AC, AT);
  k_edge_attn_wmma<<<gsz(N_EDGES / 16, T / 32), T, 0, stream>>>(B2, src, dst,
      ea_w1, ea_b1, ea_w2, ea_b2, BC, BT);
  k_scale_rows<<<gsz(NHD, T), T, 0, stream>>>(B2, AC, B0);  // Hc
  k_scale_rows<<<gsz(NHD, T), T, 0, stream>>>(B2, AT, B1);  // Ht

  // ================= GCN normalization =================
  k_fill_f32<<<fillGrid, T, 0, stream>>>(ES, 0.f, N_NODES);
  k_fill_f32<<<fillGrid, T, 0, stream>>>(ED, 0.f, N_NODES);
  k_deg<<<gsz(E2, T), T, 0, stream>>>(dst, BC, ES);
  k_deg<<<gsz(E2, T), T, 0, stream>>>(dst, BT, ED);
  k_dis<<<gsz(N_NODES, T), T, 0, stream>>>(ES);
  k_dis<<<gsz(N_NODES, T), T, 0, stream>>>(ED);

  // ================= GCN c branch =================
  k_gemm_bf16_nc64<<<gemmGrid, T, 0, stream>>>(B0, gc_w, B3, N_NODES, HD);
  k_fill_f32<<<fillGrid, T, 0, stream>>>(B4, 0.f, NHD);
  k_gcn_scatter<<<gsz(E2HD, T), T, 0, stream>>>(src, dst, BC, ES, B3, B4);
  k_bias_act<<<gsz(NHD, T), T, 0, stream>>>(B4, gc_b, 1);  // Gc

  // ================= GCN t branch =================
  k_gemm_bf16_nc64<<<gemmGrid, T, 0, stream>>>(B1, gt_w, B3, N_NODES, HD);
  k_fill_f32<<<fillGrid, T, 0, stream>>>(B2, 0.f, NHD);    // reuse H buffer as Gt
  k_gcn_scatter<<<gsz(E2HD, T), T, 0, stream>>>(src, dst, BT, ED, B3, B2);
  k_bias_act<<<gsz(NHD, T), T, 0, stream>>>(B2, gt_b, 1);  // Gt

  // ================= readouts =================
  k_gemm_bf16_nc64<<<gemmGrid, T, 0, stream>>>(B4, rc_w, B3, N_NODES, HD);
  k_fill_f32<<<fillGrid, T, 0, stream>>>(hGc, 0.f, (long long)G_GRAPHS * HD);
  k_readout<<<gsz(NHD, T), T, 0, stream>>>(B3, rc_b, batch, hGc);

  k_gemm_bf16_nc64<<<gemmGrid, T, 0, stream>>>(B2, rt_w, B3, N_NODES, HD);
  k_fill_f32<<<fillGrid, T, 0, stream>>>(hGt, 0.f, (long long)G_GRAPHS * HD);
  k_readout<<<gsz(NHD, T), T, 0, stream>>>(B3, rt_b, batch, hGt);

  // ================= classifier heads =================
  k_mlp_head<<<1, G_GRAPHS, 0, stream>>>(hGc, cc_w1, cc_b1, cc_w2, cc_b2, zc);
  k_mlp_head<<<1, G_GRAPHS, 0, stream>>>(hGt, ct_w1, ct_b1, ct_w2, ct_b2, zt);
  k_mlp_comb<<<1, G_GRAPHS, 0, stream>>>(hGc, hGt, cm_w1, cm_b1, cm_w2, cm_b2, zp);
}
